// NNFMLossContra_28269474742709
// MI455X (gfx1250) — compile-verified, matrix operand order checked by reading, not run on hardware
//
#include <hip/hip_runtime.h>
#include <hip/hip_bf16.h>
#include <cstdint>
#include <cstddef>

#define EPSF 1e-8f
#define CSEG 768
#define NPIX 9216
#define CJ   2304   // 3 * 768 joint channels

typedef __attribute__((ext_vector_type(16))) __bf16 v16bf;
typedef __attribute__((ext_vector_type(8)))  float  v8f;

struct Bytes32 { uint4 lo, hi; };

// ---------------------------------------------------------------------------
// Fragment load for V_WMMA_F32_16X16X32_BF16 from pixel-major (P x CJ) bf16.
// ISA 16-bit A 16x32 layout: lanes 0-15 -> M=lane, K = {0..7, 16..23};
// lanes 16-31 -> M=lane-16, K = {8..15, 24..31}. Per lane: two 16B chunks.
// B (32x16) loaded with the symmetric per-column layout from the transposed
// (pixel-major) store, so the same addressing applies.
// ---------------------------------------------------------------------------
__device__ __forceinline__ v16bf load_frag(const __hip_bfloat16* __restrict__ base,
                                           int row, int kBase) {
  const int lane = threadIdx.x & 31;
  const int r    = row + (lane & 15);
  const int koff = (lane >> 4) << 3;            // 0 or 8
  const __hip_bfloat16* p = base + (size_t)r * CJ + kBase + koff;
  Bytes32 s;
  s.lo = *(const uint4*)(p);                    // K chunk 0
  s.hi = *(const uint4*)(p + 16);               // K chunk +16
  return __builtin_bit_cast(v16bf, s);
}

__device__ __forceinline__ unsigned ordF(float v) {
  unsigned u = __float_as_uint(v);
  return (u & 0x80000000u) ? ~u : (u | 0x80000000u);   // monotonic float->uint
}

// ---------------------------------------------------------------------------
// K0: zero the packed argmax/argmin keys
// ---------------------------------------------------------------------------
__global__ void k_init(unsigned long long* __restrict__ bestKey,
                       unsigned long long* __restrict__ worstKey) {
  int i = blockIdx.x * blockDim.x + threadIdx.x;
  if (i < NPIX) { bestKey[i] = 0ull; worstKey[i] = 0ull; }
}

// ---------------------------------------------------------------------------
// K1: column norms (joint + per-segment) and bf16 pixel-major packing of
//     normalized joint content (At) and joint style (Bt).
//     Block = 256 threads <-> 32 pixels (lane) x 8 channel groups (grp).
// ---------------------------------------------------------------------------
__global__ void __launch_bounds__(256)
k_norm_pack(const float* __restrict__ xf, const float* __restrict__ sf,
            const float* __restrict__ rd, const float* __restrict__ sd,
            const float* __restrict__ re, const float* __restrict__ se,
            __hip_bfloat16* __restrict__ At, __hip_bfloat16* __restrict__ Bt,
            float* __restrict__ segC, float* __restrict__ segS) {
  const float* cArr[3] = { xf, rd, re };
  const float* sArr[3] = { sf, sd, se };
  const int lane = threadIdx.x & 31;
  const int grp  = threadIdx.x >> 5;          // 0..7
  const int p    = blockIdx.x * 32 + lane;    // pixel

  __shared__ float red[8][32];
  __shared__ float segRes[6][32];
  __shared__ float invC_s[32], invS_s[32];
  __shared__ __hip_bfloat16 tile[32][33];     // [channel][pixel], padded

  // ---- pass 1: per-segment sum of squares (coalesced reads) ----
  for (int s = 0; s < 6; ++s) {
    const float* a = (s < 3) ? cArr[s] : sArr[s - 3];
    float acc = 0.f;
    for (int k = grp; k < CSEG; k += 8) {
      float v = a[(size_t)k * NPIX + p];
      acc = fmaf(v, v, acc);
    }
    red[grp][lane] = acc;
    __syncthreads();
    if (grp == 0) {
      float t = 0.f;
      #pragma unroll
      for (int g = 0; g < 8; ++g) t += red[g][lane];
      segRes[s][lane] = t;
    }
    __syncthreads();
  }
  if (grp == 0) {
    float cj = segRes[0][lane] + segRes[1][lane] + segRes[2][lane];
    float sj = segRes[3][lane] + segRes[4][lane] + segRes[5][lane];
    invC_s[lane] = 1.f / (sqrtf(cj + EPSF) + EPSF);   // joint content scale
    invS_s[lane] = 1.f / (sqrtf(sj + EPSF) + EPSF);   // joint style scale
    #pragma unroll
    for (int s = 0; s < 3; ++s) {
      segC[(size_t)s * NPIX + p] = segRes[s][lane];
      segS[(size_t)s * NPIX + p] = segRes[s + 3][lane];
    }
  }
  __syncthreads();

  // ---- pass 2: normalized bf16, transposed through LDS to pixel-major ----
  for (int half = 0; half < 2; ++half) {
    const float* const* arr = (half == 0) ? cArr : sArr;
    __hip_bfloat16* dst = (half == 0) ? At : Bt;
    const float* invArr = (half == 0) ? invC_s : invS_s;
    for (int s = 0; s < 3; ++s) {
      const float* a = arr[s];
      for (int kb = 0; kb < CSEG; kb += 32) {
        // read: lane = pixel (coalesced across lanes), 4 channels/thread
        float inv = invArr[lane];
        #pragma unroll
        for (int u = 0; u < 4; ++u) {
          int kl = grp * 4 + u;
          float v = a[(size_t)(kb + kl) * NPIX + p] * inv;
          tile[kl][lane] = __float2bfloat16(v);
        }
        __syncthreads();
        // write: lane = channel (contiguous 64B per 32 lanes)
        #pragma unroll
        for (int u = 0; u < 4; ++u) {
          int pl = grp * 4 + u;
          size_t gp = (size_t)(blockIdx.x * 32 + pl);
          dst[gp * CJ + s * CSEG + kb + lane] = tile[lane][pl];
        }
        __syncthreads();
      }
    }
  }
}

// ---------------------------------------------------------------------------
// K2: bf16 WMMA GEMM (cossim = A_hat^T * B_hat) fused with per-row running
//     argmax / argmin. Workgroup = 8 waves; wave owns 16 rows; WG tile
//     128(M) x 128(N) per N-iteration; K step 32 (72 steps).
//     Software-pipelined: k+32 A/B fragments are fetched before the eight
//     WMMAs on the current fragments, so s_wait_loadcnt lands after a full
//     8-WMMA issue window instead of in front of every WMMA.
// ---------------------------------------------------------------------------
__global__ void __launch_bounds__(256, 1)
k_gemm_argminmax(const __hip_bfloat16* __restrict__ At,
                 const __hip_bfloat16* __restrict__ Bt,
                 unsigned long long* __restrict__ bestKey,
                 unsigned long long* __restrict__ worstKey) {
  const int lane    = threadIdx.x & 31;
  const int wid     = threadIdx.x >> 5;                 // 0..7
  const int rowBase = blockIdx.x * 128 + wid * 16;
  const int nStart  = blockIdx.y * (NPIX / 8);          // 1152-column chunk

  // running per-row candidates; row = rowBase + r + 8*(lane>>4)
  float bv[8], wv[8]; int bj[8], wj[8];
  #pragma unroll
  for (int r = 0; r < 8; ++r) { bv[r] = -3.4e38f; wv[r] = 3.4e38f; bj[r] = 0; wj[r] = 0; }

  for (int nTile = 0; nTile < NPIX / 8; nTile += 128) {
    const int colBase = nStart + nTile;

    v8f acc[8];
    #pragma unroll
    for (int n = 0; n < 8; ++n)
      #pragma unroll
      for (int q = 0; q < 8; ++q) acc[n][q] = 0.f;

    // prologue: fragments for k = 0
    v16bf aCur = load_frag(At, rowBase, 0);
    v16bf bCur[8];
    #pragma unroll
    for (int n = 0; n < 8; ++n) bCur[n] = load_frag(Bt, colBase + n * 16, 0);

    for (int kb = 0; kb < CJ; kb += 32) {
      // fetch next k-slab (clamped on the final iteration: redundant reload,
      // no divergence -> EXEC stays all-ones for WMMA)
      const int kn = (kb + 32 < CJ) ? (kb + 32) : kb;
      v16bf aNxt = load_frag(At, rowBase, kn);
      v16bf bNxt[8];
      #pragma unroll
      for (int n = 0; n < 8; ++n) bNxt[n] = load_frag(Bt, colBase + n * 16, kn);

      // scalar-side prefetch two k-slabs ahead (global_prefetch_b8)
      const int kp = (kb + 64 < CJ) ? (kb + 64) : kn;
      __builtin_prefetch(At + (size_t)(rowBase + (lane & 15)) * CJ + kp, 0, 3);

      // eight WMMAs on the current buffers hide the fetch latency above
      #pragma unroll
      for (int n = 0; n < 8; ++n)
        acc[n] = __builtin_amdgcn_wmma_f32_16x16x32_bf16(
                   false, aCur, false, bCur[n], (short)0, acc[n], false, false);

      aCur = aNxt;
      #pragma unroll
      for (int n = 0; n < 8; ++n) bCur[n] = bNxt[n];
    }

    // fold this 16x128 cossim strip into running per-row argmax/argmin.
    // C/D layout: VGPR r, lanes 0-15 -> (M=r, N=lane); lanes 16-31 -> (M=r+8).
    #pragma unroll
    for (int n = 0; n < 8; ++n) {
      #pragma unroll
      for (int r = 0; r < 8; ++r) {
        float mv = acc[n][r]; int mj = colBase + n * 16 + (lane & 15);
        float nv = mv;        int nj = mj;
        #pragma unroll
        for (int off = 1; off < 16; off <<= 1) {   // stays within 16-lane group
          float omv = __shfl_xor(mv, off, 32); int omj = __shfl_xor(mj, off, 32);
          if (omv > mv || (omv == mv && omj < mj)) { mv = omv; mj = omj; }
          float onv = __shfl_xor(nv, off, 32); int onj = __shfl_xor(nj, off, 32);
          if (onv < nv || (onv == nv && onj < nj)) { nv = onv; nj = onj; }
        }
        if (mv > bv[r] || (mv == bv[r] && mj < bj[r])) { bv[r] = mv; bj[r] = mj; }
        if (nv < wv[r] || (nv == wv[r] && nj < wj[r])) { wv[r] = nv; wj[r] = nj; }
      }
    }
  }

  if ((lane & 15) == 0) {
    const int g = lane >> 4;
    #pragma unroll
    for (int r = 0; r < 8; ++r) {
      const int row = rowBase + r + g * 8;
      unsigned long long bk =
        ((unsigned long long)ordF(bv[r]) << 32) | (unsigned long long)(0xFFFFFFFFu - (unsigned)bj[r]);
      unsigned long long wk =
        ((unsigned long long)(~ordF(wv[r])) << 32) | (unsigned long long)(0xFFFFFFFFu - (unsigned)wj[r]);
      atomicMax(&bestKey[row], bk);    // max cossim  == argmin d_mat
      atomicMax(&worstKey[row], wk);   // min cossim  == argmax d_mat
    }
  }
}

// ---------------------------------------------------------------------------
// K3: gathered per-segment cosines; wave per content pixel.
// ---------------------------------------------------------------------------
__global__ void __launch_bounds__(256)
k_cos_gather(const float* __restrict__ xf, const float* __restrict__ sf,
             const float* __restrict__ rd, const float* __restrict__ sd,
             const float* __restrict__ re, const float* __restrict__ se,
             const float* __restrict__ segC, const float* __restrict__ segS,
             const unsigned long long* __restrict__ bestKey,
             const unsigned long long* __restrict__ worstKey,
             float* __restrict__ pixOut) {
  const int lane = threadIdx.x & 31;
  const int wid  = threadIdx.x >> 5;
  const int i    = blockIdx.x * 8 + wid;

  const unsigned bj = 0xFFFFFFFFu - (unsigned)(bestKey[i]  & 0xFFFFFFFFull);
  const unsigned wj = 0xFFFFFFFFu - (unsigned)(worstKey[i] & 0xFFFFFFFFull);

  const float* cArr[3] = { xf, rd, re };
  const float* sArr[3] = { sf, sd, se };

  float contrib = 0.f;
  #pragma unroll
  for (int s = 0; s < 3; ++s) {
    const float* a = cArr[s]; const float* b = sArr[s];
    float db = 0.f, dw = 0.f;
    for (int k = lane; k < CSEG; k += 32) {
      float av = a[(size_t)k * NPIX + i];
      db = fmaf(av, b[(size_t)k * NPIX + bj], db);
      dw = fmaf(av, b[(size_t)k * NPIX + wj], dw);
    }
    #pragma unroll
    for (int off = 16; off; off >>= 1) {
      db += __shfl_xor(db, off, 32);
      dw += __shfl_xor(dw, off, 32);
    }
    float an  = sqrtf(segC[(size_t)s * NPIX + i])  + EPSF;   // _cos_d: no EPS in sqrt
    float bnB = sqrtf(segS[(size_t)s * NPIX + bj]) + EPSF;
    float bnW = sqrtf(segS[(size_t)s * NPIX + wj]) + EPSF;
    contrib += dw / (an * bnW) - db / (an * bnB);
  }
  if (lane == 0) pixOut[i] = contrib;
}

// ---------------------------------------------------------------------------
// K4: deterministic fixed-order reduction -> final scalar loss.
// ---------------------------------------------------------------------------
__global__ void k_final(const float* __restrict__ pixOut, float* __restrict__ out) {
  __shared__ float red[256];
  float s = 0.f;
  for (int i = threadIdx.x; i < NPIX; i += 256) s += pixOut[i];
  red[threadIdx.x] = s;
  __syncthreads();
  for (int w = 128; w; w >>= 1) {
    if ((int)threadIdx.x < w) red[threadIdx.x] += red[threadIdx.x + w];
    __syncthreads();
  }
  if (threadIdx.x == 0) out[0] = 6.0f + red[0] / (float)NPIX;
}

// ---------------------------------------------------------------------------
extern "C" void kernel_launch(void* const* d_in, const int* in_sizes, int n_in,
                              void* d_out, int out_size, void* d_ws, size_t ws_size,
                              hipStream_t stream) {
  (void)in_sizes; (void)n_in; (void)out_size; (void)ws_size;
  const float* xf = (const float*)d_in[0];   // x_feats        (content)
  const float* sf = (const float*)d_in[1];   // s_feats        (style)
  const float* rd = (const float*)d_in[2];   // output_depth_f (content)
  const float* sd = (const float*)d_in[3];   // style_depth_f  (style)
  const float* re = (const float*)d_in[4];   // output_edge_f  (content)
  const float* se = (const float*)d_in[5];   // style_edge_f   (style)
  float* out = (float*)d_out;

  char* ws = (char*)d_ws;
  const size_t bfBytes = (size_t)NPIX * CJ * sizeof(__hip_bfloat16);  // 42,467,328
  __hip_bfloat16* At = (__hip_bfloat16*)(ws);
  __hip_bfloat16* Bt = (__hip_bfloat16*)(ws + bfBytes);
  size_t off = 2 * bfBytes;
  float* segC = (float*)(ws + off);                 off += (size_t)3 * NPIX * sizeof(float);
  float* segS = (float*)(ws + off);                 off += (size_t)3 * NPIX * sizeof(float);
  unsigned long long* bestKey  = (unsigned long long*)(ws + off); off += (size_t)NPIX * 8;
  unsigned long long* worstKey = (unsigned long long*)(ws + off); off += (size_t)NPIX * 8;
  float* pixOut = (float*)(ws + off);

  k_init<<<(NPIX + 255) / 256, 256, 0, stream>>>(bestKey, worstKey);
  k_norm_pack<<<NPIX / 32, 256, 0, stream>>>(xf, sf, rd, sd, re, se, At, Bt, segC, segS);
  k_gemm_argminmax<<<dim3(NPIX / 128, 8), 256, 0, stream>>>(At, Bt, bestKey, worstKey);
  k_cos_gather<<<NPIX / 8, 256, 0, stream>>>(xf, sf, rd, sd, re, se,
                                             segC, segS, bestKey, worstKey, pixOut);
  k_final<<<1, 256, 0, stream>>>(pixOut, out);
}
